// AVWGCN_69672959475916
// MI455X (gfx1250) — compile-verified, matrix operand order checked by reading, not run on hardware
//
#include <hip/hip_runtime.h>

#define DEVINL __device__ __forceinline__

typedef __attribute__((ext_vector_type(16))) _Float16 v16h;
typedef __attribute__((ext_vector_type(8)))  _Float16 v8h;
typedef __attribute__((ext_vector_type(8)))  float    v8f;

constexpr int Bb    = 64;
constexpr int Nn    = 2048;
constexpr int DIN   = 64;
constexpr int DOUT  = 64;
constexpr int CHEBK = 3;
constexpr int EMBED = 16;
constexpr int BD    = Bb * DIN;      // 4096 (flattened batch*channel)
constexpr int KDIM  = CHEBK * DIN;   // 192  (per-node GEMM inner dim)

// ---------------------------------------------------------------------------
// Fragment loaders. CDNA5 16-bit A/B operand layout (ISA 7.12.2): lane<16
// holds K in {0..7, 16..23}, lane>=16 holds K in {8..15, 24..31}, two f16 per
// VGPR. So per lane a fragment is two contiguous 8xf16 (16 B) runs at k0 and
// k0+16, with k0 = ktile + 8*(lane>=16).
// ---------------------------------------------------------------------------
DEVINL v16h load_frag16(const _Float16* p) {
  const v8h lo = *(const v8h*)(p);
  const v8h hi = *(const v8h*)(p + 16);
  v16h r;
#pragma unroll
  for (int e = 0; e < 8; ++e) { r[e] = lo[e]; r[e + 8] = hi[e]; }
  return r;
}

DEVINL v16h load_frag_f32(const float* p) {
  v16h r;
#pragma unroll
  for (int e = 0; e < 8; ++e) {
    r[e]     = (_Float16)p[e];
    r[e + 8] = (_Float16)p[e + 16];
  }
  return r;
}

// ---------------------------------------------------------------------------
// Kernel 1: S1 = softmax(relu(E E^T)) row-wise; emit S1 (row-major f16) and
// S1^T (for use as a WMMA B operand). One block per row.
// ---------------------------------------------------------------------------
__global__ void __launch_bounds__(256)
adj_softmax_kernel(const float* __restrict__ E, _Float16* __restrict__ S1,
                   _Float16* __restrict__ S1T) {
  __shared__ float row[Nn];
  __shared__ float red[256];
  const int n = blockIdx.x, tid = threadIdx.x;

  float e[EMBED];
#pragma unroll
  for (int d = 0; d < EMBED; ++d) e[d] = E[n * EMBED + d];

  float lmax = -3.4e38f;
  for (int m = tid; m < Nn; m += 256) {
    float dot = 0.f;
#pragma unroll
    for (int d = 0; d < EMBED; ++d) dot += e[d] * E[m * EMBED + d];
    dot = fmaxf(dot, 0.f);
    row[m] = dot;
    lmax = fmaxf(lmax, dot);
  }
  red[tid] = lmax;
  __syncthreads();
  for (int s = 128; s > 0; s >>= 1) {
    if (tid < s) red[tid] = fmaxf(red[tid], red[tid + s]);
    __syncthreads();
  }
  const float mx = red[0];
  __syncthreads();

  float lsum = 0.f;
  for (int m = tid; m < Nn; m += 256) {
    const float p = __expf(row[m] - mx);
    row[m] = p;
    lsum += p;
  }
  red[tid] = lsum;
  __syncthreads();
  for (int s = 128; s > 0; s >>= 1) {
    if (tid < s) red[tid] += red[tid + s];
    __syncthreads();
  }
  const float inv = 1.f / red[0];

  for (int m = tid; m < Nn; m += 256) {
    const _Float16 h = (_Float16)(row[m] * inv);
    S1[(size_t)n * Nn + m]  = h;
    S1T[(size_t)m * Nn + n] = h;
  }
}

// ---------------------------------------------------------------------------
// Kernel 2: X^T (4096 x 2048, f16): xTT[b*64+c][m] = x[b][m][c].
// LDS-tiled so both global read and write are coalesced.
// ---------------------------------------------------------------------------
__global__ void __launch_bounds__(256)
xt_transpose_kernel(const float* __restrict__ x, _Float16* __restrict__ xTT) {
  __shared__ float tile[64][65];
  const int m0 = blockIdx.x * 64;
  const int b  = blockIdx.y;
  for (int idx = threadIdx.x; idx < 64 * 64; idx += 256) {
    const int mi = idx >> 6, c = idx & 63;
    tile[mi][c] = x[((size_t)b * Nn + (m0 + mi)) * DIN + c];
  }
  __syncthreads();
  for (int idx = threadIdx.x; idx < 64 * 64; idx += 256) {
    const int c = idx >> 6, mi = idx & 63;
    xTT[((size_t)(b * DIN + c)) * Nn + (m0 + mi)] = (_Float16)tile[mi][c];
  }
}

// ---------------------------------------------------------------------------
// Kernel 3: generic f16 WMMA GEMM, C = alpha * A @ B (- I optionally).
//   A : row-major M x K (lda), BT : row-major N x K (ldb) == B^T,
//   C : row-major M x N (ldc), f16 output, f32 accumulation.
// 128 threads = 4 waves arranged 2x2; 64x64 output per wave (16 v8f accums),
// 128x128 per block, K stepped by 32. Data reuse via 192 MB L2.
// ---------------------------------------------------------------------------
__global__ void __launch_bounds__(128)
wmma_gemm_f16(const _Float16* __restrict__ A, const _Float16* __restrict__ BT,
              _Float16* __restrict__ C, int K, int lda, int ldb, int ldc,
              float alpha, int subDiag) {
  const int tid  = threadIdx.x;
  const int wave = tid >> 5, lane = tid & 31;
  const int half = lane >> 4, l16 = lane & 15;
  const int tileM = blockIdx.y * 128 + (wave >> 1) * 64;
  const int tileN = blockIdx.x * 128 + (wave & 1) * 64;

  v8f acc[4][4] = {};
  for (int kt = 0; kt < K; kt += 32) {
    v16h af[4], bf[4];
#pragma unroll
    for (int mi = 0; mi < 4; ++mi)
      af[mi] = load_frag16(A + (size_t)(tileM + mi * 16 + l16) * lda + kt + 8 * half);
#pragma unroll
    for (int ni = 0; ni < 4; ++ni)
      bf[ni] = load_frag16(BT + (size_t)(tileN + ni * 16 + l16) * ldb + kt + 8 * half);
#pragma unroll
    for (int mi = 0; mi < 4; ++mi)
#pragma unroll
      for (int ni = 0; ni < 4; ++ni)
        acc[mi][ni] = __builtin_amdgcn_wmma_f32_16x16x32_f16(
            false, af[mi], false, bf[ni], (short)0, acc[mi][ni], false, false);
  }

  // Epilogue. C/D layout: VGPR g -> row = g + 8*(lane>=16), col = lane%16.
#pragma unroll
  for (int mi = 0; mi < 4; ++mi)
#pragma unroll
    for (int ni = 0; ni < 4; ++ni)
#pragma unroll
      for (int g = 0; g < 8; ++g) {
        const int grow = tileM + mi * 16 + g + 8 * half;
        const int gcol = tileN + ni * 16 + l16;
        float v = acc[mi][ni][g] * alpha;
        if (subDiag && grow == gcol) v -= 1.f;  // T2 = 2*S*S - I
        C[(size_t)grow * ldc + gcol] = (_Float16)v;
      }
}

// ---------------------------------------------------------------------------
// Kernel 4: per-node fused weight generation + (64 x 192) @ (192 x 64) GEMM.
// W_n = E_n @ W_pool computed into LDS pre-transposed (WT[o][k*64+i], padded
// row stride 200 f16 = 400 B, 16-B aligned) so B fragments are ds_load_b128.
// A rows: k=0 term streamed from original f32 x; k=1,2 from xg1/xg2 (f16).
// One block (4 waves) per node; wave w covers batch rows [16w, 16w+16).
// ---------------------------------------------------------------------------
__global__ void __launch_bounds__(128)
node_gemm_kernel(const float* __restrict__ E, const float* __restrict__ Wp,
                 const float* __restrict__ biasP, const float* __restrict__ x,
                 const _Float16* __restrict__ xg1, const _Float16* __restrict__ xg2,
                 float* __restrict__ out) {
  constexpr int WROW = 200;  // KDIM(192) + 8 pad
  __shared__ _Float16 WT[DOUT * WROW];
  __shared__ float bsh[DOUT];
  const int n = blockIdx.x, tid = threadIdx.x;

  float e[EMBED];
#pragma unroll
  for (int d = 0; d < EMBED; ++d) e[d] = E[n * EMBED + d];

  for (int j = tid; j < KDIM * DOUT; j += 128) {  // j = k*4096 + i*64 + o
    float acc = 0.f;
#pragma unroll
    for (int d = 0; d < EMBED; ++d) acc += e[d] * Wp[d * (KDIM * DOUT) + j];
    WT[(j & 63) * WROW + (j >> 6)] = (_Float16)acc;  // WT[o][k*64+i]
  }
  if (tid < DOUT) {
    float acc = 0.f;
#pragma unroll
    for (int d = 0; d < EMBED; ++d) acc += e[d] * biasP[d * DOUT + tid];
    bsh[tid] = acc;
  }
  __syncthreads();

  const int wave = tid >> 5, lane = tid & 31;
  const int half = lane >> 4, l16 = lane & 15;
  const int row = wave * 16 + l16;  // batch index for A fragment

  v8f acc[4] = {};
  for (int kt = 0; kt < KDIM; kt += 32) {   // kt uniform -> EXEC stays full
    const int kc = kt >> 6;                  // Chebyshev order
    const int i0 = (kt & 63) + 8 * half;     // channel offset within order
    v16h af;
    if (kc == 0) {
      af = load_frag_f32(x + ((size_t)row * Nn + n) * DIN + i0);
    } else {
      const _Float16* src = (kc == 1) ? xg1 : xg2;
      af = load_frag16(src + (size_t)n * BD + row * DIN + i0);
    }
#pragma unroll
    for (int ni = 0; ni < 4; ++ni) {
      const v16h bf = load_frag16(&WT[(ni * 16 + l16) * WROW + kt + 8 * half]);
      acc[ni] = __builtin_amdgcn_wmma_f32_16x16x32_f16(
          false, af, false, bf, (short)0, acc[ni], false, false);
    }
  }

#pragma unroll
  for (int ni = 0; ni < 4; ++ni)
#pragma unroll
    for (int g = 0; g < 8; ++g) {
      const int b = wave * 16 + g + 8 * half;
      const int o = ni * 16 + l16;
      out[((size_t)b * Nn + n) * DOUT + o] = acc[ni][g] + bsh[o];
    }
}

// ---------------------------------------------------------------------------
// Host-side launch. Inputs: x(B,N,64) f32, E(N,16) f32, W_pool(16,3,64,64) f32,
// bias_pool(16,64) f32. Output: (B,N,64) f32.
// Workspace layout (f16): S1 8.39MB | S1T 8.39MB | T2 8.39MB |
//                         X^T 16.78MB | xg1 16.78MB | xg2 16.78MB  (75.5 MB)
// ---------------------------------------------------------------------------
extern "C" void kernel_launch(void* const* d_in, const int* in_sizes, int n_in,
                              void* d_out, int out_size, void* d_ws, size_t ws_size,
                              hipStream_t stream) {
  const float* x  = (const float*)d_in[0];
  const float* E  = (const float*)d_in[1];
  const float* Wp = (const float*)d_in[2];
  const float* bp = (const float*)d_in[3];
  float* out = (float*)d_out;

  const size_t szS  = (size_t)Nn * Nn * sizeof(_Float16);  // 8,388,608
  const size_t szXG = (size_t)Nn * BD * sizeof(_Float16);  // 16,777,216
  char* ws = (char*)d_ws;
  _Float16* S1  = (_Float16*)(ws);
  _Float16* S1T = (_Float16*)(ws + szS);
  _Float16* T2  = (_Float16*)(ws + 2 * szS);
  _Float16* xTT = (_Float16*)(ws + 3 * szS);
  _Float16* xg1 = (_Float16*)(ws + 3 * szS + szXG);
  _Float16* xg2 = (_Float16*)(ws + 3 * szS + 2 * szXG);

  // 1) S1 = softmax(relu(E E^T)), plus transposed copy for B-operand use.
  adj_softmax_kernel<<<Nn, 256, 0, stream>>>(E, S1, S1T);

  // 2) X^T in f16 for use as B operand of the spatial GEMMs.
  xt_transpose_kernel<<<dim3(Nn / 64, Bb), 256, 0, stream>>>(x, xTT);

  // 3) T2 = 2 * S1 @ S1 - I   (M=N=K=2048)
  wmma_gemm_f16<<<dim3(Nn / 128, Nn / 128), 128, 0, stream>>>(
      S1, S1T, T2, Nn, Nn, Nn, Nn, 2.0f, 1);

  // 4) xg1 = S1 @ X  (M=2048, N=4096, K=2048)
  wmma_gemm_f16<<<dim3(BD / 128, Nn / 128), 128, 0, stream>>>(
      S1, xTT, xg1, Nn, Nn, Nn, BD, 1.0f, 0);

  // 5) xg2 = T2 @ X
  wmma_gemm_f16<<<dim3(BD / 128, Nn / 128), 128, 0, stream>>>(
      T2, xTT, xg2, Nn, Nn, Nn, BD, 1.0f, 0);

  // 6) out[b,n,:] = [x | xg1 | xg2]_n @ W_n + bias_n  (per-node fused GEMM)
  node_gemm_kernel<<<Nn, 128, 0, stream>>>(E, Wp, bp, x, xg1, xg2, out);
}
